// MDA_33870112096825
// MI455X (gfx1250) — matrix operand order repaired and checked by reading, not compiled, not run on hardware
//
#include <hip/hip_runtime.h>
#include <math.h>

typedef __attribute__((ext_vector_type(16))) _Float16 v16h;
typedef __attribute__((ext_vector_type(8)))  _Float16 v8h;
typedef __attribute__((ext_vector_type(8)))  float    v8f;

#define KPAD 320        // 310 padded to multiple of 32
#define KCH  10         // KPAD / 32 k-chunks for the cosine GEMM
#define COLS_PER_BUF 32 // tgt columns staged per LDS buffer (2 WMMA tiles)
#define LDS_STRIDE 328  // halfs per staged column (320 + 8 pad, 16B aligned)

// sched_group_barrier masks (LLVM AMDGPUIGroupLP)
#define SGB_WMMA    0x008
#define SGB_DS_READ 0x100

// ---------------------------------------------------------------------------
// prep_rows: one wave per row. fp32 -> f16 with zero pad to KPAD, plus the
// RECIPROCAL norm 1/max(||row||, 1e-4)  (so ns*nt >= 1e-8 ~ the EPS guard,
// and the cosine epilogue needs only multiplies, no division chain).
// ---------------------------------------------------------------------------
__global__ void prep_rows(const float* __restrict__ src, const float* __restrict__ tgt,
                          _Float16* __restrict__ Sh, _Float16* __restrict__ Th,
                          float* __restrict__ nsi, float* __restrict__ nti,
                          int N, int M, int F)
{
    const int lane = threadIdx.x & 31;
    const int wave = (blockIdx.x * blockDim.x + threadIdx.x) >> 5;
    if (wave >= N + M) return;
    const float* row; _Float16* outp; float* nrm;
    if (wave < N) { row = src + (size_t)wave * F; outp = Sh + (size_t)wave * KPAD; nrm = nsi + wave; }
    else { int r = wave - N; row = tgt + (size_t)r * F; outp = Th + (size_t)r * KPAD; nrm = nti + r; }
    float ss = 0.f;
    for (int k = lane; k < KPAD; k += 32) {
        float v = (k < F) ? row[k] : 0.f;
        outp[k] = (_Float16)v;
        ss += v * v;
    }
    for (int off = 16; off; off >>= 1) ss += __shfl_xor(ss, off, 32);
    if (lane == 0) *nrm = 1.0f / fmaxf(sqrtf(ss), 1e-4f);
}

// ---------------------------------------------------------------------------
// prep_wt: transpose fp32 weight [fin][fout] -> f16 WT [Npad][Kpad], zero pad.
// ---------------------------------------------------------------------------
__global__ void prep_wt(const float* __restrict__ W, _Float16* __restrict__ WT,
                        int fin, int fout, int Kp, int Np)
{
    int idx = blockIdx.x * blockDim.x + threadIdx.x;
    if (idx >= Kp * Np) return;
    int n = idx / Kp, k = idx % Kp;
    WT[idx] = (n < fout && k < fin) ? (_Float16)W[(size_t)k * fout + n] : (_Float16)0.f;
}

// ---------------------------------------------------------------------------
// Async-stage one group of 32 tgt columns (each 320 halfs = 40 x 16B chunks)
// into an LDS buffer laid out [col][LDS_STRIDE]. 256 threads x 5 chunks each.
// Uses gfx1250 GLOBAL_LOAD_ASYNC_TO_LDS_B128 (tracked by ASYNCcnt).
// ---------------------------------------------------------------------------
__device__ __forceinline__ void stage_async(const _Float16* __restrict__ Th,
                                            _Float16* dst, int g, int tid)
{
    const _Float16* gsrc = Th + (size_t)g * COLS_PER_BUF * KPAD;
#pragma unroll
    for (int i = 0; i < 5; ++i) {
        const int idx = tid + i * 256;          // 0..1279
        const int col = idx / 40;
        const int kc  = idx % 40;               // 16B chunk within the column
        unsigned lds_addr = (unsigned)(size_t)(dst + col * LDS_STRIDE) + (unsigned)(kc * 16);
        unsigned long long gaddr =
            (unsigned long long)(size_t)(gsrc + (size_t)col * KPAD) + (unsigned long long)(kc * 16);
        asm volatile("global_load_async_to_lds_b128 %0, %1, off"
                     :: "v"(lds_addr), "v"(gaddr) : "memory");
    }
}

// ---------------------------------------------------------------------------
// cos_argmax: 8 waves/block, each wave owns one 16-row stripe (A fragments
// hoisted into VGPRs once). All waves share LDS-staged tgt tiles, double-
// buffered via async copies. Two independent WMMA accumulator chains are
// interleaved; a sched_group_barrier pipeline (DS-read 4 ahead of each WMMA
// pair) forces the scheduler to keep ds_load_b128s in flight under the WMMAs.
// Grid must be exactly Mrows/128 blocks (EXEC stays all-ones for WMMA).
// ---------------------------------------------------------------------------
__global__ __launch_bounds__(256)
void cos_argmax(const _Float16* __restrict__ Sh, const _Float16* __restrict__ Th,
                const float* __restrict__ nsi, const float* __restrict__ nti,
                int* __restrict__ topIdx, int Mrows, int Ncols)
{
    __shared__ _Float16 buf[2][COLS_PER_BUF * LDS_STRIDE];
    const int tid   = threadIdx.x;
    const int lane  = tid & 31;
    const int wv    = tid >> 5;
    const int rBase = (blockIdx.x * 8 + wv) * 16;
    const int rowA  = rBase + (lane & 15);
    const int kOffA = (lane >> 4) << 3;        // 0 or 8
    const int col16 = lane & 15;
    const int kOffB = (lane >> 4) << 3;        // 0 or 8

    // Hoist A fragments for this 16-row stripe (16 x KPAD)
    v16h a[KCH];
#pragma unroll
    for (int c = 0; c < KCH; ++c) {
        v8h lo = *(const v8h*)(Sh + (size_t)rowA * KPAD + c * 32 + kOffA);
        v8h hi = *(const v8h*)(Sh + (size_t)rowA * KPAD + c * 32 + 16 + kOffA);
        a[c] = __builtin_shufflevector(lo, hi, 0,1,2,3,4,5,6,7,8,9,10,11,12,13,14,15);
    }
    float nsr[8];                              // reciprocal src norms
#pragma unroll
    for (int r = 0; r < 8; ++r) nsr[r] = nsi[rBase + r + ((lane >> 4) << 3)];

    float bestV[8]; int bestI[8];
#pragma unroll
    for (int r = 0; r < 8; ++r) { bestV[r] = -3.0e38f; bestI[r] = 0; }

    const int nGroups = Ncols / COLS_PER_BUF;   // 256
    stage_async(Th, &buf[0][0], 0, tid);
    asm volatile("s_wait_asynccnt 0" ::: "memory");
    __syncthreads();

    for (int g = 0; g < nGroups; ++g) {
        const int pb = g & 1;
        if (g + 1 < nGroups) stage_async(Th, &buf[pb ^ 1][0], g + 1, tid);

        const _Float16* bbase = &buf[pb][0];
        const _Float16* bp0 = bbase + col16 * LDS_STRIDE + kOffB;
        const _Float16* bp1 = bbase + (16 + col16) * LDS_STRIDE + kOffB;
        v8f acc0 = {}, acc1 = {};
#pragma unroll
        for (int c = 0; c < KCH; ++c) {
            v16h b0 = __builtin_shufflevector(*(const v8h*)(bp0 + c * 32), *(const v8h*)(bp0 + c * 32 + 16), 0,1,2,3,4,5,6,7,8,9,10,11,12,13,14,15);
            v16h b1 = __builtin_shufflevector(*(const v8h*)(bp1 + c * 32), *(const v8h*)(bp1 + c * 32 + 16), 0,1,2,3,4,5,6,7,8,9,10,11,12,13,14,15);
            acc0 = __builtin_amdgcn_wmma_f32_16x16x32_f16(
                false, a[c], false, b0, (short)0, acc0, false, false);
            acc1 = __builtin_amdgcn_wmma_f32_16x16x32_f16(
                false, a[c], false, b1, (short)0, acc1, false, false);
        }
        // Desired pipeline for this scheduling region: one chunk's DS reads
        // (4x ds_load_b128) run ahead, then alternate WMMA pair / next chunk's
        // DS reads. Forces overlapping live ranges -> distinct B registers ->
        // partial s_wait_dscnt instead of full drains.
        __builtin_amdgcn_sched_group_barrier(SGB_DS_READ, 4, 0);
#pragma unroll
        for (int c = 0; c < KCH - 1; ++c) {
            __builtin_amdgcn_sched_group_barrier(SGB_WMMA, 2, 0);
            __builtin_amdgcn_sched_group_barrier(SGB_DS_READ, 4, 0);
        }
        __builtin_amdgcn_sched_group_barrier(SGB_WMMA, 2, 0);

        const int   ci0  = g * COLS_PER_BUF + col16;
        const int   ci1  = ci0 + 16;
        const float nti0 = nti[ci0];
        const float nti1 = nti[ci1];
#pragma unroll
        for (int r = 0; r < 8; ++r) {
            float c0 = acc0[r] * (nsr[r] * nti0);
            float c1 = acc1[r] * (nsr[r] * nti1);
            if (c0 > bestV[r]) { bestV[r] = c0; bestI[r] = ci0; }
            if (c1 > bestV[r]) { bestV[r] = c1; bestI[r] = ci1; }
        }
        // next buffer staged + this buffer fully consumed before overwrite
        asm volatile("s_wait_asynccnt 0" ::: "memory");
        __syncthreads();
    }

    // Reduce (max, min-index tiebreak) across the 16 lanes sharing each row
    for (int off = 1; off < 16; off <<= 1) {
#pragma unroll
        for (int r = 0; r < 8; ++r) {
            float ov = __shfl_xor(bestV[r], off, 32);
            int   oi = __shfl_xor(bestI[r], off, 32);
            if (ov > bestV[r] || (ov == bestV[r] && oi < bestI[r])) { bestV[r] = ov; bestI[r] = oi; }
        }
    }
    if ((lane & 15) == 0) {
        const int mBase = rBase + ((lane >> 4) << 3);
#pragma unroll
        for (int r = 0; r < 8; ++r) topIdx[mBase + r] = bestI[r];
    }
    (void)Mrows;
}

// ---------------------------------------------------------------------------
// mix_rows: feature_mix = 0.8*src + 0.2*tgt[top]; rows <32 keep raw src.
// ---------------------------------------------------------------------------
__global__ void mix_rows(const float* __restrict__ src, const float* __restrict__ tgt,
                         const int* __restrict__ top, _Float16* __restrict__ Mh,
                         int N, int F)
{
    size_t idx = (size_t)blockIdx.x * blockDim.x + threadIdx.x;
    if (idx >= (size_t)N * KPAD) return;
    int r = (int)(idx / KPAD), k = (int)(idx % KPAD);
    float v = 0.f;
    if (k < F) {
        float s = src[(size_t)r * F + k];
        v = (r < 32) ? s : 0.8f * s + 0.2f * tgt[(size_t)top[r] * F + k];
    }
    Mh[idx] = (_Float16)v;
}

// ---------------------------------------------------------------------------
// Generic WMMA GEMM, templated on K-chunks so fragment arrays stay in VGPRs
// and the WMMA chain issues back-to-back after a single load clause.
// out = act((A[M x KC*32] @ WT^T + bias) * scale); one 16x16 tile per wave.
// ---------------------------------------------------------------------------
template<int KC>
__global__ __launch_bounds__(256)
void wmma_gemm(const _Float16* __restrict__ A, const _Float16* __restrict__ WT,
               const float* __restrict__ bias,
               _Float16* __restrict__ outH, float* __restrict__ outF,
               int Mrows, int Npad, int Nvalid, int leaky, float scale)
{
    constexpr int Kpad = KC * 32;
    const int lane = threadIdx.x & 31;
    const int wave = (blockIdx.x * blockDim.x + threadIdx.x) >> 5;
    const int tilesN = Npad >> 4;
    if (wave >= (Mrows >> 4) * tilesN) return;   // wave-uniform
    const int tm = wave / tilesN, tn = wave % tilesN;
    const int rBase = tm << 4, cBase = tn << 4;
    const int rowA  = rBase + (lane & 15);
    const int kOffA = (lane >> 4) << 3;
    const int colB  = cBase + (lane & 15);
    const int kOffB = (lane >> 4) << 3;

    const _Float16* ap = A  + (size_t)rowA * Kpad + kOffA;
    const _Float16* bp = WT + (size_t)colB * Kpad + kOffB;

    v16h afr[KC], bfr[KC];
#pragma unroll
    for (int c = 0; c < KC; ++c) {
        v8h lo = *(const v8h*)(ap + c * 32);
        v8h hi = *(const v8h*)(ap + c * 32 + 16);
        afr[c] = __builtin_shufflevector(lo, hi, 0,1,2,3,4,5,6,7,8,9,10,11,12,13,14,15);
        bfr[c] = __builtin_shufflevector(*(const v8h*)(bp + c * 32), *(const v8h*)(bp + c * 32 + 16), 0,1,2,3,4,5,6,7,8,9,10,11,12,13,14,15);
    }
    v8f acc = {};
#pragma unroll
    for (int c = 0; c < KC; ++c)
        acc = __builtin_amdgcn_wmma_f32_16x16x32_f16(
            false, afr[c], false, bfr[c], (short)0, acc, false, false);

    const int n = cBase + (lane & 15);
    const float bval = (n < Nvalid) ? bias[n] : 0.f;
#pragma unroll
    for (int r = 0; r < 8; ++r) {
        int m = rBase + r + ((lane >> 4) << 3);
        float v = (acc[r] + bval) * scale;
        if (leaky) v = (v > 0.f) ? v : 0.01f * v;
        size_t o = (size_t)m * Npad + n;
        if (outH) outH[o] = (_Float16)v;
        if (outF) outF[o] = v;
    }
}

// ---------------------------------------------------------------------------
// Loss helpers (match reference's log_softmax(softmax(x)) quirk exactly)
// ---------------------------------------------------------------------------
__device__ __forceinline__ float dsm_ce(const float* x, int n, int sel)
{
    float mx = x[0];
    for (int j = 1; j < n; ++j) mx = fmaxf(mx, x[j]);
    float Z = 0.f;
    for (int j = 0; j < n; ++j) Z += expf(x[j] - mx);
    float invZ = 1.f / Z;
    float mx2 = 0.f, psel = 0.f;
    for (int j = 0; j < n; ++j) {
        float p = expf(x[j] - mx) * invZ;
        mx2 = fmaxf(mx2, p);
        if (j == sel) psel = p;
    }
    float Z2 = 0.f;
    for (int j = 0; j < n; ++j) Z2 += expf(expf(x[j] - mx) * invZ - mx2);
    return -(psel - (mx2 + logf(Z2)));
}

__device__ __forceinline__ float dcond_ent(const float* x, int n)
{
    float mx = x[0];
    for (int j = 1; j < n; ++j) mx = fmaxf(mx, x[j]);
    float Z = 0.f;
    for (int j = 0; j < n; ++j) Z += expf(x[j] - mx);
    float lse = mx + logf(Z);
    float s = 0.f;
    for (int j = 0; j < n; ++j) { float lp = x[j] - lse; s += expf(lp) * lp; }
    return -s;
}

__global__ void loss_rows(const float* __restrict__ disf, const float* __restrict__ psf,
                          const float* __restrict__ ptf, const int* __restrict__ label,
                          const int* __restrict__ markp,
                          float* __restrict__ rvCls, float* __restrict__ rvDom,
                          float* __restrict__ rvCe, int N)
{
    int r = blockIdx.x * blockDim.x + threadIdx.x;
    if (r >= N) return;
    int mark = markp[0];
    rvDom[r] = dsm_ce(disf + (size_t)r * 16, 14, mark);
    rvCls[r] = dsm_ce(psf  + (size_t)r * 16, 4, label[r]);
    rvCe[r]  = dcond_ent(ptf + (size_t)r * 16, 4);
}

__global__ void mmd_col(const _Float16* __restrict__ bs, const _Float16* __restrict__ bt,
                        float* __restrict__ s, int N)
{
    __shared__ float sm[256];
    const int j = blockIdx.x;           // 0..31
    float acc = 0.f;
    for (int r = threadIdx.x; r < N; r += 256)
        acc += (float)bs[(size_t)r * 32 + j] - (float)bt[(size_t)r * 32 + j];
    sm[threadIdx.x] = acc; __syncthreads();
    for (int o = 128; o; o >>= 1) { if (threadIdx.x < o) sm[threadIdx.x] += sm[threadIdx.x + o]; __syncthreads(); }
    if (threadIdx.x == 0) s[j] = sm[0];
}

__global__ void finalize_losses(const float* __restrict__ rvCls, const float* __restrict__ rvDom,
                                const float* __restrict__ rvCe, const float* __restrict__ s32,
                                float* __restrict__ out, int N)
{
    __shared__ float sm[256];
    const int t = threadIdx.x;
    float a = 0.f, b = 0.f, c = 0.f;
    for (int r = t; r < N; r += 256) { a += rvCls[r]; b += rvDom[r]; c += rvCe[r]; }
    sm[t] = a; __syncthreads();
    for (int o = 128; o; o >>= 1) { if (t < o) sm[t] += sm[t + o]; __syncthreads(); }
    if (t == 0) out[0] = sm[0] / (float)N;
    __syncthreads();
    sm[t] = b; __syncthreads();
    for (int o = 128; o; o >>= 1) { if (t < o) sm[t] += sm[t + o]; __syncthreads(); }
    if (t == 0) out[2] = sm[0] / (float)N;
    __syncthreads();
    sm[t] = c; __syncthreads();
    for (int o = 128; o; o >>= 1) { if (t < o) sm[t] += sm[t + o]; __syncthreads(); }
    if (t == 0) {
        out[3] = sm[0] / (float)N;
        float d = 0.f;
        for (int j = 0; j < 32; ++j) d += s32[j] * s32[j];
        out[1] = d / ((float)N * (float)N);
    }
}

// ---------------------------------------------------------------------------
extern "C" void kernel_launch(void* const* d_in, const int* in_sizes, int n_in,
                              void* d_out, int out_size, void* d_ws, size_t ws_size,
                              hipStream_t stream)
{
    (void)in_sizes; (void)n_in; (void)out_size; (void)ws_size;
    const int N = 8192, M = 8192, F = 310;

    const float* src  = (const float*)d_in[0];
    const float* tgt  = (const float*)d_in[1];
    const float* w1   = (const float*)d_in[2];   const float* b1 = (const float*)d_in[3];
    const float* w2   = (const float*)d_in[4];   const float* b2 = (const float*)d_in[5];
    const float* w3   = (const float*)d_in[6];   const float* b3 = (const float*)d_in[7];
    const float* dw1  = (const float*)d_in[8];   const float* db1 = (const float*)d_in[9];
    const float* dw2  = (const float*)d_in[10];  const float* db2 = (const float*)d_in[11];
    const float* bw   = (const float*)d_in[12];  const float* bb = (const float*)d_in[13];
    const float* cw   = (const float*)d_in[14];  const float* cb = (const float*)d_in[15];
    const int* label  = (const int*)d_in[16];
    const int* markp  = (const int*)d_in[17];
    float* out = (float*)d_out;

    char* ws = (char*)d_ws;
    size_t off = 0;
    auto carve = [&](size_t bytes) -> char* {
        char* p = ws + off;
        off = (off + bytes + 255) & ~(size_t)255;
        return p;
    };
    _Float16* Sh   = (_Float16*)carve((size_t)N * KPAD * 2);
    _Float16* Th   = (_Float16*)carve((size_t)M * KPAD * 2);
    _Float16* Mix  = (_Float16*)carve((size_t)N * KPAD * 2);
    float*    nsrc = (float*)carve((size_t)N * 4);
    float*    ntgt = (float*)carve((size_t)M * 4);
    int*      top  = (int*)carve((size_t)N * 4);
    _Float16* H1s  = (_Float16*)carve((size_t)N * 256 * 2);
    _Float16* H1t  = (_Float16*)carve((size_t)M * 256 * 2);
    _Float16* H2s  = (_Float16*)carve((size_t)N * 128 * 2);
    _Float16* H2t  = (_Float16*)carve((size_t)M * 128 * 2);
    _Float16* C3s  = (_Float16*)carve((size_t)N * 64 * 2);
    _Float16* C3t  = (_Float16*)carve((size_t)M * 64 * 2);
    _Float16* D1   = (_Float16*)carve((size_t)N * 32 * 2);
    float*    DisF = (float*)carve((size_t)N * 16 * 4);
    _Float16* Bs   = (_Float16*)carve((size_t)N * 32 * 2);
    _Float16* Bt   = (_Float16*)carve((size_t)M * 32 * 2);
    float*    PsF  = (float*)carve((size_t)N * 16 * 4);
    float*    PtF  = (float*)carve((size_t)M * 16 * 4);
    float*    rvC  = (float*)carve((size_t)N * 4);
    float*    rvD  = (float*)carve((size_t)N * 4);
    float*    rvE  = (float*)carve((size_t)N * 4);
    float*    Svec = (float*)carve(32 * 4);
    _Float16* WT1  = (_Float16*)carve((size_t)256 * KPAD * 2);
    _Float16* WT2  = (_Float16*)carve((size_t)128 * 256 * 2);
    _Float16* WT3  = (_Float16*)carve((size_t)64 * 128 * 2);
    _Float16* WTd1 = (_Float16*)carve((size_t)32 * 64 * 2);
    _Float16* WTd2 = (_Float16*)carve((size_t)16 * 32 * 2);
    _Float16* WTb  = (_Float16*)carve((size_t)32 * 64 * 2);
    _Float16* WTc  = (_Float16*)carve((size_t)16 * 32 * 2);

    // --- stage 0: convert rows to f16 + reciprocal norms; weights to f16 ---
    {
        int waves = N + M;
        prep_rows<<<(waves * 32 + 255) / 256, 256, 0, stream>>>(src, tgt, Sh, Th, nsrc, ntgt, N, M, F);
    }
    auto wprep = [&](const float* W, _Float16* WT, int fin, int fout, int Kp, int Np) {
        int tot = Kp * Np;
        prep_wt<<<(tot + 255) / 256, 256, 0, stream>>>(W, WT, fin, fout, Kp, Np);
    };
    wprep(w1,  WT1,  310, 256, KPAD, 256);
    wprep(w2,  WT2,  256, 128, 256, 128);
    wprep(w3,  WT3,  128, 64,  128, 64);
    wprep(dw1, WTd1, 64,  32,  64,  32);
    wprep(dw2, WTd2, 32,  14,  32,  16);
    wprep(bw,  WTb,  64,  32,  64,  32);
    wprep(cw,  WTc,  32,  4,   32,  16);

    // --- stage 1: fused cosine GEMM + argmax (WMMA + async LDS staging) ---
    cos_argmax<<<N / 128, 256, 0, stream>>>(Sh, Th, nsrc, ntgt, top, N, M);

    // --- stage 2: hybrid feature mix ---
    {
        size_t tot = (size_t)N * KPAD;
        mix_rows<<<(unsigned)((tot + 255) / 256), 256, 0, stream>>>(src, tgt, top, Mix, N, F);
    }

    // --- stage 3: MLP stack via templated WMMA GEMM ---
    auto gemm = [&](const _Float16* A, const _Float16* WT, const float* bias,
                    _Float16* oh, float* of, int Mr, int Kp, int Np, int Nv,
                    int leaky, float scale) {
        int waves = (Mr / 16) * (Np / 16);
        int blocks = (waves + 7) / 8;
        switch (Kp) {
            case 320: wmma_gemm<10><<<blocks, 256, 0, stream>>>(A, WT, bias, oh, of, Mr, Np, Nv, leaky, scale); break;
            case 256: wmma_gemm<8> <<<blocks, 256, 0, stream>>>(A, WT, bias, oh, of, Mr, Np, Nv, leaky, scale); break;
            case 128: wmma_gemm<4> <<<blocks, 256, 0, stream>>>(A, WT, bias, oh, of, Mr, Np, Nv, leaky, scale); break;
            case 64:  wmma_gemm<2> <<<blocks, 256, 0, stream>>>(A, WT, bias, oh, of, Mr, Np, Nv, leaky, scale); break;
            default:  wmma_gemm<1> <<<blocks, 256, 0, stream>>>(A, WT, bias, oh, of, Mr, Np, Nv, leaky, scale); break;
        }
    };
    gemm(Mix, WT1, b1, H1s, nullptr, N, KPAD, 256, 256, 1, 1.f);
    gemm(Th,  WT1, b1, H1t, nullptr, M, KPAD, 256, 256, 1, 1.f);
    gemm(H1s, WT2, b2, H2s, nullptr, N, 256, 128, 128, 1, 1.f);
    gemm(H1t, WT2, b2, H2t, nullptr, M, 256, 128, 128, 1, 1.f);
    gemm(H2s, WT3, b3, C3s, nullptr, N, 128, 64, 64, 1, 1.f);
    gemm(H2t, WT3, b3, C3t, nullptr, M, 128, 64, 64, 1, 1.f);
    gemm(C3s, WTd1, db1, D1, nullptr, N, 64, 32, 32, 1, 1.f);
    gemm(D1,  WTd2, db2, nullptr, DisF, N, 32, 16, 14, 0, 0.8f);   // * LAMDA
    gemm(C3s, WTb, bb, Bs, nullptr, N, 64, 32, 32, 1, 1.f);
    gemm(C3t, WTb, bb, Bt, nullptr, M, 64, 32, 32, 1, 1.f);
    gemm(Bs,  WTc, cb, nullptr, PsF, N, 32, 16, 4, 0, 1.f);
    gemm(Bt,  WTc, cb, nullptr, PtF, M, 32, 16, 4, 0, 1.f);

    // --- stage 4: losses (deterministic fixed-order reductions) ---
    loss_rows<<<N / 256, 256, 0, stream>>>(DisF, PsF, PtF, label, markp, rvC, rvD, rvE, N);
    mmd_col<<<32, 256, 0, stream>>>(Bs, Bt, Svec, N);
    finalize_losses<<<1, 256, 0, stream>>>(rvC, rvD, rvE, Svec, out, N);
}
